// SingleHeadAttention_48249662603713
// MI455X (gfx1250) — matrix operand order repaired and checked by reading, compiled, not run on hardware
//
#include <hip/hip_runtime.h>
#include <hip/hip_bf16.h>

typedef __attribute__((ext_vector_type(16))) _Float16 v16h;
typedef __attribute__((ext_vector_type(8)))  _Float16 v8h;
typedef __attribute__((ext_vector_type(4)))  _Float16 v4h;
typedef __attribute__((ext_vector_type(8)))  float    v8f;
typedef __attribute__((ext_vector_type(4)))  float    v4f;
typedef __attribute__((ext_vector_type(4)))  unsigned int v4u;
typedef __attribute__((ext_vector_type(8)))  int      v8i;
typedef __attribute__((ext_vector_type(4)))  int      v4i;

#define WMMA_F16(a, b, c) \
  __builtin_amdgcn_wmma_f32_16x16x32_f16(false, (a), false, (b), (short)0, (c), false, false)

static constexpr int B_ = 8;
static constexpr int S_ = 2048;
static constexpr int D_ = 1024;
static constexpr float SCALE_ = 0.03125f;   // 1/sqrt(1024)
static constexpr float NEG_FILL_ = -1e10f;

// Padded LDS A-tile geometry (TDM inserts 16B after every 1024B chunk):
// row stride = 2048 + 2*16 = 2080 bytes = 1040 halves; pad = 8 halves per chunk.
static constexpr int XROW_H = 1040;

// ---------------------------------------------------------------------------
// Kernel 0: convert x fp32 -> f16 once (removes 192x redundant conversions).
// ---------------------------------------------------------------------------
__global__ __launch_bounds__(256) void sha_cvtx_kernel(
    const float* __restrict__ x, _Float16* __restrict__ xh) {
  const size_t i4 = ((size_t)blockIdx.x * 256 + threadIdx.x) * 4;
  v4f v = *(const v4f*)(x + i4);
  v4h o;
#pragma unroll
  for (int i = 0; i < 4; ++i) o[i] = (_Float16)v[i];
  *(v4h*)(xh + i4) = o;
}

// ---------------------------------------------------------------------------
// Kernel 1: Wq/Wk/Wv (fp32 [k][n]) -> WT (f16 [z][n][k]): transposed so every
// WMMA B-fragment load is two contiguous global_load_b128.
// ---------------------------------------------------------------------------
__global__ __launch_bounds__(256) void sha_cvtw_kernel(
    const float* __restrict__ Wq, const float* __restrict__ Wk,
    const float* __restrict__ Wv, _Float16* __restrict__ WT) {
  const int idx = blockIdx.x * 256 + threadIdx.x;   // over D_*D_
  const int z = blockIdx.y;
  const float* W = (z == 0) ? Wq : (z == 1) ? Wk : Wv;
  const int n = idx >> 10;
  const int k = idx & 1023;
  WT[(size_t)z * D_ * D_ + idx] = (_Float16)W[(size_t)k * D_ + n];
}

// ---------------------------------------------------------------------------
// Kernel 2: QKV projection GEMM.
//  - Block = 8 waves = one 16-row tile x all 1024 columns of one matrix.
//  - A tile (16x1024 f16, 32KB+pad) staged into LDS ONCE per block by the
//    Tensor Data Mover (tensor_load_to_lds, D# built per ISA ch.8), with LDS
//    padding so A-fragment ds_load_b128s don't all hit bank 0.
//  - Each wave owns a 16x128 strip: one A fragment feeds 8 WMMAs per k-step.
// ---------------------------------------------------------------------------
__global__ __launch_bounds__(256) void sha_qkv_kernel(
    const _Float16* __restrict__ xh, const _Float16* __restrict__ WT,
    _Float16* __restrict__ Qh, _Float16* __restrict__ Kh,
    _Float16* __restrict__ VT) {
  __shared__ _Float16 s_x[16 * XROW_H];   // 33280 B

  const int lane = threadIdx.x & 31;
  const int wave = threadIdx.x >> 5;
  const int h  = lane >> 4;
  const int ln = lane & 15;
  const int z    = blockIdx.z;            // 0:Q 1:K 2:V
  const int row0 = blockIdx.x * 16;       // flattened B*S row tile
  const int colw = wave * 128;            // this wave's column strip

  if (wave == 0) {
    // ---- TDM descriptor: 2D tile, 1024x16 halves, row stride 1024 ---------
    const unsigned lds_off =
        (unsigned)(size_t)(__attribute__((address_space(3))) _Float16*)s_x;
    const unsigned long long ga =
        (unsigned long long)(const void*)(xh + (size_t)row0 * D_);
    v4u g0;
    g0[0] = 1u;                                   // count=1 (valid user D#)
    g0[1] = lds_off;                              // lds_addr (bytes)
    g0[2] = (unsigned)(ga & 0xFFFFFFFFu);         // global_addr[31:0]
    g0[3] = (unsigned)((ga >> 32) & 0x01FFFFFFu)  // global_addr[56:32]
            | (2u << 30);                         // type = 2 ("image")
    v8i g1;
    g1[0] = (1 << 16)      // data_size = 1 -> 2 bytes
          | (1 << 20)      // pad_enable
          | (7 << 22)      // pad_interval = 7 -> every 256 dwords (1024 B)
          | (3 << 25);     // pad_amount = 3 -> 4 dwords (16 B)
    g1[1] = (int)(1024u << 16);   // tensor_dim0[15:0]=1024 in [31:16]
    g1[2] = (int)(16u << 16);     // tensor_dim0 hi=0; tensor_dim1[15:0]=16
    g1[3] = (int)(1024u << 16);   // tensor_dim1 hi=0; tile_dim0=1024
    g1[4] = 16;                   // tile_dim1=16; tile_dim2=0 (2D)
    g1[5] = 1024;                 // tensor_dim0_stride lo32 = 1024 elements
    g1[6] = 0;
    g1[7] = 0;
    v4i znull4 = {};
    v8i znull8 = {};
    // 6-arg toolchain variant: (g0, g1, g2, g3, g4, cpol)
    __builtin_amdgcn_tensor_load_to_lds(g0, g1, znull4, znull4, znull8, 0);
    __builtin_amdgcn_s_wait_tensorcnt(0);
  }
  __syncthreads();

  const _Float16* wbase = WT + (size_t)z * D_ * D_ + 16 * h;
  v8f acc[8] = {};

  for (int d0 = 0; d0 < D_; d0 += 32) {
    // A fragment from padded LDS tile: lane row = ln, halves K=16h+i, 16h+8+i
    const unsigned rowh = (unsigned)ln * XROW_H;
    const unsigned e0 = (unsigned)(d0 + 16 * h);
    const unsigned e1 = e0 + 8;
    v8h alo = *(const v8h*)&s_x[rowh + e0 + ((e0 >= 512u) ? 8u : 0u)];
    v8h ahi = *(const v8h*)&s_x[rowh + e1 + ((e1 >= 512u) ? 8u : 0u)];
    v16h a;
#pragma unroll
    for (int i = 0; i < 8; ++i) { a[i] = alo[i]; a[8 + i] = ahi[i]; }
#pragma unroll
    for (int ct = 0; ct < 8; ++ct) {
      v16h b = *(const v16h*)(wbase + (size_t)(colw + ct * 16 + ln) * D_ + d0);
      acc[ct] = WMMA_F16(a, b, acc[ct]);
    }
  }

  if (z == 2) {   // V stored transposed: VT[b][col][s]
    const int bi = row0 / S_, s0 = row0 % S_;
#pragma unroll
    for (int ct = 0; ct < 8; ++ct) {
      _Float16* vt = VT + ((size_t)bi * D_ + colw + ct * 16 + ln) * S_ + s0 + 8 * h;
#pragma unroll
      for (int v = 0; v < 8; ++v) vt[v] = (_Float16)acc[ct][v];
    }
  } else {
    _Float16* dst = (z == 0) ? Qh : Kh;
#pragma unroll
    for (int ct = 0; ct < 8; ++ct)
#pragma unroll
      for (int v = 0; v < 8; ++v)
        dst[(size_t)(row0 + v + 8 * h) * D_ + colw + ct * 16 + ln] = (_Float16)acc[ct][v];
  }
}

// ---------------------------------------------------------------------------
// Kernel 3: causal flash attention. Block = 8 waves = one (batch, 16-query)
// tile. Score d-contraction split across waves (128 d each), reduced via
// ds_add_f32; threads 0-15 run online softmax; each wave owns 128 output
// columns and does P@V against VT. Q slice register-resident.
// ---------------------------------------------------------------------------
__global__ __launch_bounds__(256) void sha_attn_kernel(
    const _Float16* __restrict__ Qh, const _Float16* __restrict__ Kh,
    const _Float16* __restrict__ VT, float* __restrict__ out) {
  __shared__ float    s_sc[16 * 32];
  __shared__ _Float16 s_p[16 * 32];
  __shared__ float    s_m[16], s_l[16], s_a[16];

  const int lane = threadIdx.x & 31;
  const int wave = threadIdx.x >> 5;
  const int h  = lane >> 4;
  const int ln = lane & 15;
  const int bi  = blockIdx.y;
  const int q0  = blockIdx.x * 16;
  const int dlo = wave * 128;          // this wave's d-slice AND output columns

  if (threadIdx.x < 16) { s_m[threadIdx.x] = -__builtin_inff(); s_l[threadIdx.x] = 0.f; }

  v8f acc[8] = {};
  const _Float16* qrow  = Qh + (size_t)(bi * S_ + q0 + ln) * D_;
  const _Float16* kbase = Kh + (size_t)bi * S_ * D_;

  // Register-resident Q fragments for this wave's 128-d slice (4 x v16h).
  v16h qf[4];
#pragma unroll
  for (int t = 0; t < 4; ++t) {
    const _Float16* ap = qrow + dlo + t * 32 + 16 * h;
    v8h alo = *(const v8h*)ap;
    v8h ahi = *(const v8h*)(ap + 8);
#pragma unroll
    for (int i = 0; i < 8; ++i) { qf[t][i] = alo[i]; qf[t][8 + i] = ahi[i]; }
  }
  __syncthreads();

  for (int kb = 0; kb < q0 + 16; kb += 32) {
    for (int i = threadIdx.x; i < 16 * 32; i += 256) s_sc[i] = 0.f;
    // prefetch next key tile's rows for this d-slice (global_prefetch_b8)
    __builtin_prefetch(kbase + (size_t)(kb + 32 + ln) * D_ + dlo, 0, 1);
    __syncthreads();

    // --- partial scores over d in [dlo, dlo+128): 8 WMMAs -------------------
    v8f ps0 = {}, ps1 = {};
#pragma unroll
    for (int t = 0; t < 4; ++t) {
      const int d0 = dlo + t * 32;
      const _Float16* kp = kbase + (size_t)(kb + ln) * D_ + d0 + 16 * h;
      v16h b0 = *(const v16h*)kp;
      v16h b1 = *(const v16h*)(kp + (size_t)16 * D_);
      ps0 = WMMA_F16(qf[t], b0, ps0);
      ps1 = WMMA_F16(qf[t], b1, ps1);
    }
#pragma unroll
    for (int v = 0; v < 8; ++v) {
      const int r = v + 8 * h;
      atomicAdd(&s_sc[r * 32 + ln],      ps0[v]);   // ds_add_f32
      atomicAdd(&s_sc[r * 32 + 16 + ln], ps1[v]);
    }
    __syncthreads();

    // --- online softmax (one thread per query row) --------------------------
    if (threadIdx.x < 16) {
      const int r = threadIdx.x, qi = q0 + r;
      const float mo = s_m[r];
      float sv[32], mx = mo;
#pragma unroll
      for (int j = 0; j < 32; ++j) {
        float s = s_sc[r * 32 + j] * SCALE_;
        if (kb + j > qi) s = NEG_FILL_;              // causal mask, matches ref
        sv[j] = s;
        mx = fmaxf(mx, s);
      }
      const float alpha = __expf(mo - mx);
      float sum = 0.f;
#pragma unroll
      for (int j = 0; j < 32; ++j) {
        const float p = __expf(sv[j] - mx);
        s_p[r * 32 + j] = (_Float16)p;
        sum += p;
      }
      s_m[r] = mx;
      s_l[r] = s_l[r] * alpha + sum;
      s_a[r] = alpha;
    }
    __syncthreads();

    // --- rescale accumulator and P@V over this wave's 128 columns -----------
    v16h pa;
    {
      const _Float16* pr = &s_p[ln * 32 + 16 * h];
      v8h plo = *(const v8h*)pr;
      v8h phi = *(const v8h*)(pr + 8);
#pragma unroll
      for (int i = 0; i < 8; ++i) { pa[i] = plo[i]; pa[8 + i] = phi[i]; }
    }
    float al[8];
#pragma unroll
    for (int v = 0; v < 8; ++v) al[v] = s_a[v + 8 * h];
#pragma unroll
    for (int ct = 0; ct < 8; ++ct) {
#pragma unroll
      for (int v = 0; v < 8; ++v) acc[ct][v] *= al[v];
      const _Float16* vp = VT + ((size_t)bi * D_ + dlo + ct * 16 + ln) * S_ + kb + 16 * h;
      v16h vb = *(const v16h*)vp;
      acc[ct] = WMMA_F16(pa, vb, acc[ct]);
    }
    // next iteration's top __syncthreads protects s_p/s_a from early overwrite
  }

  float inv_l[8];
#pragma unroll
  for (int v = 0; v < 8; ++v) inv_l[v] = 1.f / s_l[v + 8 * h];
#pragma unroll
  for (int ct = 0; ct < 8; ++ct)
#pragma unroll
    for (int v = 0; v < 8; ++v)
      out[(size_t)(bi * S_ + q0 + v + 8 * h) * D_ + dlo + ct * 16 + ln] =
          acc[ct][v] * inv_l[v];
}

// ---------------------------------------------------------------------------
// Workspace layout (f16):
//   [0,    6MB) : WT (3 x 1024 x 1024)
//   [6,   38MB) : xh (16384 x 1024)
//   [38,  70MB) : Qh (16384 x 1024)
//   [70, 102MB) : Kh (16384 x 1024)
//   [102,134MB) : VT (8 x 1024 x 2048)
// ---------------------------------------------------------------------------
extern "C" void kernel_launch(void* const* d_in, const int* in_sizes, int n_in,
                              void* d_out, int out_size, void* d_ws, size_t ws_size,
                              hipStream_t stream) {
  const float* x  = (const float*)d_in[0];
  const float* Wq = (const float*)d_in[1];
  const float* Wk = (const float*)d_in[2];
  const float* Wv = (const float*)d_in[3];
  char* ws = (char*)d_ws;
  _Float16* WT = (_Float16*)ws;
  _Float16* xh = (_Float16*)(ws + (size_t)6   * 1024 * 1024);
  _Float16* Qh = (_Float16*)(ws + (size_t)38  * 1024 * 1024);
  _Float16* Kh = (_Float16*)(ws + (size_t)70  * 1024 * 1024);
  _Float16* VT = (_Float16*)(ws + (size_t)102 * 1024 * 1024);
  float* out = (float*)d_out;

  sha_cvtx_kernel<<<dim3((B_ * S_ * D_) / (4 * 256)), 256, 0, stream>>>(x, xh);
  sha_cvtw_kernel<<<dim3((D_ * D_) / 256, 3), 256, 0, stream>>>(Wq, Wk, Wv, WT);
  sha_qkv_kernel<<<dim3((B_ * S_) / 16, 1, 3), 256, 0, stream>>>(xh, WT, Qh, Kh, VT);
  sha_attn_kernel<<<dim3(S_ / 16, B_), 256, 0, stream>>>(Qh, Kh, VT, out);
}
